// DecoderWithAttention_77326591197576
// MI455X (gfx1250) — compile-verified, matrix operand order checked by reading, not run on hardware
//
#include <hip/hip_runtime.h>
#include <hip/hip_bf16.h>

// ---------------- problem dims ----------------
#define BB      256
#define N_OBJ   36
#define OBJ_D   2048
#define ATT_D   1024
#define EMB_D   1024
#define DEC_D   1024
#define VOCAB   10000
#define VOCABP  10016          /* padded to multiple of 32 for 2x2 wave tiling */
#define TT      20

// d_out layout (flat concat, reference return order)
#define PRED_OFF   0LL                                   // B*T*VOCAB = 51,200,000
#define CAP_OFF    51200000LL                            // B*T = 5120
#define LEN_OFF    51205120LL                            // B
#define ALPHA_OFF  51205376LL                            // B*T*36 = 184,320
#define SORT_OFF   51389696LL                            // B

typedef __attribute__((ext_vector_type(16))) __bf16 v16bf;
typedef __attribute__((ext_vector_type(8)))  __bf16 bf16x8;
typedef __attribute__((ext_vector_type(8)))  float  v8f;

// ---------------- bf16 helpers (bit-level, no __bf16 arithmetic) ----------------
__device__ __forceinline__ float bfu2f(unsigned short h) {
    return __uint_as_float(((unsigned int)h) << 16);
}
__device__ __forceinline__ unsigned short f2bfu(float f) {
    unsigned int u = __float_as_uint(f);
    unsigned int r = u + 0x7FFFu + ((u >> 16) & 1u);   // round-to-nearest-even
    return (unsigned short)(r >> 16);
}
__device__ __forceinline__ float sigmoidf_(float x) {
    return 1.0f / (1.0f + __expf(-x));
}

// ---------------- gfx1250 async global->LDS helpers ----------------
// GLOBAL_LOAD_ASYNC_TO_LDS_B128: per-lane LDS dest offset in vdst VGPR,
// 64-bit global address in VGPR pair; tracked by ASYNCcnt (ISA 15.18.3 op 98).
__device__ __forceinline__ void async_ld_lds_b128(unsigned lds_off, const void* gaddr) {
    asm volatile("global_load_async_to_lds_b128 %0, %1, off"
                 :: "v"(lds_off), "v"(gaddr) : "memory");
}
__device__ __forceinline__ void wait_asynccnt0() {
    asm volatile("s_wait_asynccnt 0" ::: "memory");
}

// ---------------- stable descending sort by caption length ----------------
__global__ void sort_kernel(const int* __restrict__ cap_len,   // (B,1)
                            const int* __restrict__ caps,      // (B,T)
                            int* __restrict__ sort_ind,
                            int* __restrict__ len_sorted,
                            int* __restrict__ cap_sorted) {
    int b = threadIdx.x;
    if (b >= BB) return;
    int myLen = cap_len[b];
    int rank = 0;
    for (int j = 0; j < BB; ++j) {
        int lj = cap_len[j];
        if (lj > myLen || (lj == myLen && j < b)) rank++;
    }
    sort_ind[rank]   = b;
    len_sorted[rank] = myLen;
    for (int t = 0; t < TT; ++t) cap_sorted[rank * TT + t] = caps[b * TT + t];
}

// small int outputs -> float d_out regions
__global__ void misc_out(const int* __restrict__ cap_sorted,
                         const int* __restrict__ len_sorted,
                         const int* __restrict__ sort_ind,
                         float* __restrict__ out) {
    int i = blockIdx.x * 256 + threadIdx.x;
    if (i < BB * TT)                 out[CAP_OFF  + i]              = (float)cap_sorted[i];
    else if (i < BB * TT + BB)       out[LEN_OFF  + (i - BB * TT)]  = (float)len_sorted[i - BB * TT];
    else if (i < BB * TT + 2 * BB)   out[SORT_OFF + (i - BB * TT - BB)] = (float)sort_ind[i - BB * TT - BB];
}

// ---------------- conversions ----------------
// converts src[0..n_src) to bf16, zero-fills [n_src..n_total) (row padding)
__global__ void f2bf_kernel(const float* __restrict__ s, unsigned short* __restrict__ d,
                            long long n_src, long long n_total) {
    long long i = (long long)blockIdx.x * 256 + threadIdx.x;
    if (i >= n_total) return;
    d[i] = (i < n_src) ? f2bfu(s[i]) : (unsigned short)0;
}

__global__ void build_wcat(const float* __restrict__ wih,   // (4096,3072)
                           const float* __restrict__ whh,   // (4096,1024)
                           unsigned short* __restrict__ wcat) { // (4096,4096)
    long long i = (long long)blockIdx.x * 256 + threadIdx.x;
    if (i >= 4096LL * 4096LL) return;
    int n = (int)(i >> 12), k = (int)(i & 4095);
    float v = (k < 3072) ? wih[(size_t)n * 3072 + k] : whh[(size_t)n * 1024 + (k - 3072)];
    wcat[i] = f2bfu(v);
}

__global__ void add_bias2(const float* __restrict__ a, const float* __restrict__ b,
                          float* __restrict__ d, int n) {
    int i = blockIdx.x * 256 + threadIdx.x;
    if (i < n) d[i] = a[i] + b[i];
}

__global__ void gather_obj(const float* __restrict__ obj, const int* __restrict__ sort_ind,
                           unsigned short* __restrict__ out) {
    long long i = (long long)blockIdx.x * 256 + threadIdx.x;
    const long long per = (long long)N_OBJ * OBJ_D;      // 73728
    if (i >= (long long)BB * per) return;
    int b = (int)(i / per);
    long long r = i - (long long)b * per;
    out[i] = f2bfu(obj[(long long)sort_ind[b] * per + r]);
}

__global__ void build_stepx(const float* __restrict__ enc,       // (B,1024)
                            const float* __restrict__ emb,       // (VOCAB,1024)
                            const int* __restrict__ sort_ind,
                            const int* __restrict__ cap_sorted,  // (B,T)
                            unsigned short* __restrict__ out) {  // (B,T,1024)
    long long i = (long long)blockIdx.x * 256 + threadIdx.x;
    if (i >= (long long)BB * TT * EMB_D) return;
    int b = (int)(i / (TT * EMB_D));
    int r = (int)(i % (TT * EMB_D));
    int t = r >> 10, k = r & 1023;
    float v;
    if (t == 0) v = enc[(size_t)sort_ind[b] * EMB_D + k];
    else        v = emb[(size_t)cap_sorted[b * TT + t - 1] * EMB_D + k];
    out[i] = f2bfu(v);
}

__global__ void init_hc(float* __restrict__ h, float* __restrict__ c,
                        unsigned short* __restrict__ h_bf) {
    int i = blockIdx.x * 256 + threadIdx.x;
    if (i < BB * DEC_D) { h[i] = 0.0f; c[i] = 0.0f; h_bf[i] = 0; }
}

// ---------------- WMMA bf16 GEMM with LDS-shared weights ----------------
// C[M,N] = A[M,K] * B[N,K]^T + bias.
// One 256-thread block = 8 waves; block owns a 256-row M panel x one 32-wide N
// strip. The 32x32 bf16 weight chunk per K-step is staged once into LDS with
// GLOBAL_LOAD_ASYNC_TO_LDS_B128 (double buffered, ASYNCcnt + barrier), then all
// 8 waves read their B fragments from LDS -> 8x less L2 weight traffic (the
// dominant term at M=256). Each wave computes a 2x2 grid of 16x16 WMMA tiles.
// LDS rows padded to 40 elems (20 banks) -> conflict-free ds_load_b128 frags.
#define LDSW 40
__global__ void gemm_wmma_bf16(const unsigned short* __restrict__ Au, int lda,
                               const unsigned short* __restrict__ Bu, int ldb,
                               const float* __restrict__ bias,
                               float* __restrict__ C, long long ldc,
                               int M, int N, int Npad, int K,
                               const int* __restrict__ rowlen, int t) {
    __shared__ unsigned short Btile[2][32][LDSW];

    const __bf16* A = reinterpret_cast<const __bf16*>(Au);
    int tid  = threadIdx.x;
    int lane = tid & 31;
    int wave = tid >> 5;

    int stripsN  = Npad >> 5;                 // 32-wide N strips
    int mPanels  = M >> 8;                    // 256-row M panels
    int nstrip   = blockIdx.x / mPanels;
    int mpanel   = blockIdx.x - nstrip * mPanels;
    int tn = nstrip << 5;
    int tm = (mpanel << 8) + (wave << 5);

    int rlo   = lane & 15;
    int khalf = (lane & 16) ? 8 : 0;

    const __bf16* arow0 = A + (size_t)(tm + rlo)      * lda + khalf;
    const __bf16* arow1 = A + (size_t)(tm + 16 + rlo) * lda + khalf;

    // staging map: threads 0..127 each move one 16B chunk of the 32x32 chunk
    int srow  = tid >> 2;          // 0..31
    int scol8 = (tid & 3) << 3;    // 0,8,16,24
    const unsigned short* Bg = Bu + (size_t)(tn + srow) * ldb + scol8;

    // prologue: stage k=0 into buffer 0
    if (tid < 128)
        async_ld_lds_b128((unsigned)(size_t)&Btile[0][srow][scol8], Bg);
    wait_asynccnt0();
    __syncthreads();

    v8f acc00 = {}, acc01 = {}, acc10 = {}, acc11 = {};
    for (int k = 0; k < K; k += 32) {
        int buf = (k >> 5) & 1;
        if (k + 32 < K && tid < 128)
            async_ld_lds_b128((unsigned)(size_t)&Btile[buf ^ 1][srow][scol8], Bg + k + 32);

        __builtin_prefetch(arow0 + k + 128, 0, 1);   // global_prefetch_b8
        __builtin_prefetch(arow1 + k + 128, 0, 1);

        bf16x8 a00 = *reinterpret_cast<const bf16x8*>(arow0 + k);
        bf16x8 a01 = *reinterpret_cast<const bf16x8*>(arow0 + k + 16);
        bf16x8 a10 = *reinterpret_cast<const bf16x8*>(arow1 + k);
        bf16x8 a11 = *reinterpret_cast<const bf16x8*>(arow1 + k + 16);
        bf16x8 b00 = *reinterpret_cast<const bf16x8*>(&Btile[buf][rlo][khalf]);
        bf16x8 b01 = *reinterpret_cast<const bf16x8*>(&Btile[buf][rlo][khalf + 16]);
        bf16x8 b10 = *reinterpret_cast<const bf16x8*>(&Btile[buf][rlo + 16][khalf]);
        bf16x8 b11 = *reinterpret_cast<const bf16x8*>(&Btile[buf][rlo + 16][khalf + 16]);

        v16bf af0 = __builtin_shufflevector(a00, a01, 0,1,2,3,4,5,6,7,8,9,10,11,12,13,14,15);
        v16bf af1 = __builtin_shufflevector(a10, a11, 0,1,2,3,4,5,6,7,8,9,10,11,12,13,14,15);
        v16bf bf0 = __builtin_shufflevector(b00, b01, 0,1,2,3,4,5,6,7,8,9,10,11,12,13,14,15);
        v16bf bf1 = __builtin_shufflevector(b10, b11, 0,1,2,3,4,5,6,7,8,9,10,11,12,13,14,15);

        acc00 = __builtin_amdgcn_wmma_f32_16x16x32_bf16(false, af0, false, bf0,
                                                        (short)0, acc00, false, false);
        acc01 = __builtin_amdgcn_wmma_f32_16x16x32_bf16(false, af0, false, bf1,
                                                        (short)0, acc01, false, false);
        acc10 = __builtin_amdgcn_wmma_f32_16x16x32_bf16(false, af1, false, bf0,
                                                        (short)0, acc10, false, false);
        acc11 = __builtin_amdgcn_wmma_f32_16x16x32_bf16(false, af1, false, bf1,
                                                        (short)0, acc11, false, false);

        if (k + 32 < K) wait_asynccnt0();   // own async writes done; barrier covers peers
        __syncthreads();
    }

    int mhi  = (lane & 16) ? 8 : 0;
    int col0 = tn + rlo;
    int col1 = tn + 16 + rlo;
    float bv0 = (bias && col0 < N) ? bias[col0] : 0.0f;
    float bv1 = (bias && col1 < N) ? bias[col1] : 0.0f;
#pragma unroll
    for (int r = 0; r < 8; ++r) {
        int row0 = tm + r + mhi;
        int row1 = tm + 16 + r + mhi;
        bool z0 = rowlen && (t >= rowlen[row0]);
        bool z1 = rowlen && (t >= rowlen[row1]);
        if (col0 < N) {
            C[(size_t)row0 * ldc + col0] = z0 ? 0.0f : (acc00[r] + bv0);
            C[(size_t)row1 * ldc + col0] = z1 ? 0.0f : (acc10[r] + bv0);
        }
        if (col1 < N) {
            C[(size_t)row0 * ldc + col1] = z0 ? 0.0f : (acc01[r] + bv1);
            C[(size_t)row1 * ldc + col1] = z1 ? 0.0f : (acc11[r] + bv1);
        }
    }
}

// ---------------- attention scores: e[b,n] = relu(att1+att2) . w_full + b_full ----------------
__global__ void att_score(const float* __restrict__ att1,   // (B*36,1024)
                          const float* __restrict__ att2,   // (B,1024)
                          const float* __restrict__ wfull,  // (1024)
                          const float* __restrict__ bfull,  // (1)
                          float* __restrict__ e) {          // (B*36)
    int gw   = (blockIdx.x * blockDim.x + threadIdx.x) >> 5;
    int lane = threadIdx.x & 31;
    if (gw >= BB * N_OBJ) return;
    int b = gw / N_OBJ;
    const float* a1 = att1 + (size_t)gw * ATT_D;
    const float* a2 = att2 + (size_t)b * ATT_D;
    float s = 0.0f;
    for (int k = lane; k < ATT_D; k += 32) {
        float v = a1[k] + a2[k];
        v = v > 0.0f ? v : 0.0f;
        s += v * wfull[k];
    }
    for (int off = 16; off; off >>= 1) s += __shfl_xor(s, off, 32);
    if (lane == 0) e[gw] = s + bfull[0];
}

// ---------------- softmax over 36 objects + masked alphas output ----------------
__global__ void softmax_alpha(const float* __restrict__ e,
                              float* __restrict__ alpha,
                              float* __restrict__ alphas_out,  // d_out + ALPHA_OFF
                              const int* __restrict__ len_sorted, int t) {
    int b = threadIdx.x;
    if (b >= BB) return;
    const float* eb = e + b * N_OBJ;
    float m = -1e30f;
    for (int n = 0; n < N_OBJ; ++n) m = fmaxf(m, eb[n]);
    float s = 0.0f;
    for (int n = 0; n < N_OBJ; ++n) s += __expf(eb[n] - m);
    float inv = 1.0f / s;
    bool act = t < len_sorted[b];
    for (int n = 0; n < N_OBJ; ++n) {
        float a = __expf(eb[n] - m) * inv;
        alpha[b * N_OBJ + n] = a;
        alphas_out[(size_t)b * (TT * N_OBJ) + t * N_OBJ + n] = act ? a : 0.0f;
    }
}

// ---------------- awe = alpha@objects, gate, build xcat=[x_t, gate*awe, h] (bf16) ----------------
__global__ void awe_gate_xcat(const float* __restrict__ alpha,          // (B,36)
                              const unsigned short* __restrict__ obj_bf,// (B,36,2048)
                              const float* __restrict__ fgate,          // (B,2048) pre-sigmoid
                              const unsigned short* __restrict__ stepx, // (B,T,1024)
                              const unsigned short* __restrict__ h_bf,  // (B,1024)
                              unsigned short* __restrict__ xh_bf,       // (B,4096)
                              int t) {
    int idx = blockIdx.x * 256 + threadIdx.x;
    if (idx >= BB * 4096) return;
    int b = idx >> 12, o = idx & 4095;
    unsigned short out;
    if (o < 1024) {
        out = stepx[((size_t)b * TT + t) * EMB_D + o];
    } else if (o < 3072) {
        int oo = o - 1024;
        const float* al = alpha + b * N_OBJ;
        const unsigned short* ob = obj_bf + (size_t)b * N_OBJ * OBJ_D + oo;
        float s = 0.0f;
#pragma unroll 4
        for (int n = 0; n < N_OBJ; ++n) s += al[n] * bfu2f(ob[(size_t)n * OBJ_D]);
        float g = sigmoidf_(fgate[(size_t)b * OBJ_D + oo]);
        out = f2bfu(g * s);
    } else {
        out = h_bf[(size_t)b * DEC_D + (o - 3072)];
    }
    xh_bf[idx] = out;
}

// ---------------- LSTM pointwise + masked state update ----------------
__global__ void lstm_pw(const float* __restrict__ gates,     // (B,4096)
                        float* __restrict__ h, float* __restrict__ c,
                        unsigned short* __restrict__ h_bf,
                        unsigned short* __restrict__ hnew_bf,
                        const int* __restrict__ len_sorted, int t) {
    int idx = blockIdx.x * 256 + threadIdx.x;
    if (idx >= BB * DEC_D) return;
    int b = idx >> 10, d = idx & 1023;
    const float* g = gates + (size_t)b * 4096;
    float si = sigmoidf_(g[d]);
    float sf = sigmoidf_(g[1024 + d]);
    float gg = tanhf(g[2048 + d]);
    float so = sigmoidf_(g[3072 + d]);
    float cn = sf * c[idx] + si * gg;
    float hn = so * tanhf(cn);
    bool act = t < len_sorted[b];
    float hv = act ? hn : h[idx];
    float cv = act ? cn : c[idx];
    h[idx] = hv;
    c[idx] = cv;
    h_bf[idx]    = f2bfu(hv);
    hnew_bf[idx] = f2bfu(hn);   // preds use unmasked h_new
}

// ---------------- host side ----------------
static inline void launch_gemm(const void* A, int lda, const void* B, int ldb,
                               const float* bias, float* C, long long ldc,
                               int M, int N, int Npad, int K,
                               const int* rowlen, int t, hipStream_t s) {
    int blocks = (M / 256) * (Npad / 32);
    gemm_wmma_bf16<<<blocks, 256, 0, s>>>((const unsigned short*)A, lda,
                                          (const unsigned short*)B, ldb,
                                          bias, C, ldc, M, N, Npad, K, rowlen, t);
}

extern "C" void kernel_launch(void* const* d_in, const int* in_sizes, int n_in,
                              void* d_out, int out_size, void* d_ws, size_t ws_size,
                              hipStream_t stream) {
    const float* encoder_out = (const float*)d_in[0];
    const float* objects     = (const float*)d_in[1];
    const int*   enc_caps    = (const int*)  d_in[2];
    const int*   cap_lens    = (const int*)  d_in[3];
    const float* emb_table   = (const float*)d_in[4];
    const float* W_enc_att   = (const float*)d_in[5];
    const float* b_enc_att   = (const float*)d_in[6];
    const float* W_dec_att   = (const float*)d_in[7];
    const float* b_dec_att   = (const float*)d_in[8];
    const float* w_full_att  = (const float*)d_in[9];
    const float* b_full_att  = (const float*)d_in[10];
    const float* W_fbeta     = (const float*)d_in[11];
    const float* b_fbeta     = (const float*)d_in[12];
    const float* W_ih        = (const float*)d_in[13];
    const float* W_hh        = (const float*)d_in[14];
    const float* b_ih        = (const float*)d_in[15];
    const float* b_hh        = (const float*)d_in[16];
    const float* W_fc        = (const float*)d_in[17];
    const float* b_fc        = (const float*)d_in[18];

    float* out = (float*)d_out;

    // workspace carve-up (256B aligned)
    char* ws = (char*)d_ws;
    size_t off = 0;
    auto alloc = [&](size_t bytes) -> void* {
        void* p = ws + off;
        off = (off + bytes + 255) & ~(size_t)255;
        return p;
    };
    int*            sort_ind   = (int*)           alloc(BB * 4);
    int*            len_sorted = (int*)           alloc(BB * 4);
    int*            cap_sorted = (int*)           alloc(BB * TT * 4);
    unsigned short* Wenc_bf    = (unsigned short*)alloc((size_t)ATT_D * OBJ_D * 2);
    unsigned short* Wdec_bf    = (unsigned short*)alloc((size_t)ATT_D * DEC_D * 2);
    unsigned short* Wfb_bf     = (unsigned short*)alloc((size_t)OBJ_D * DEC_D * 2);
    unsigned short* Wcat_bf    = (unsigned short*)alloc((size_t)4096 * 4096 * 2);
    unsigned short* Wfc_bf     = (unsigned short*)alloc((size_t)VOCABP * DEC_D * 2);  // padded rows
    float*          bcat       = (float*)         alloc(4096 * 4);
    unsigned short* obj_bf     = (unsigned short*)alloc((size_t)BB * N_OBJ * OBJ_D * 2);
    unsigned short* stepx_bf   = (unsigned short*)alloc((size_t)BB * TT * EMB_D * 2);
    float*          att1       = (float*)         alloc((size_t)BB * N_OBJ * ATT_D * 4);
    float*          att2       = (float*)         alloc((size_t)BB * ATT_D * 4);
    float*          evec       = (float*)         alloc((size_t)BB * N_OBJ * 4);
    float*          alpha      = (float*)         alloc((size_t)BB * N_OBJ * 4);
    float*          fgate      = (float*)         alloc((size_t)BB * OBJ_D * 4);
    unsigned short* xh_bf      = (unsigned short*)alloc((size_t)BB * 4096 * 2);
    float*          gatesum    = (float*)         alloc((size_t)BB * 4096 * 4);
    float*          hbuf       = (float*)         alloc((size_t)BB * DEC_D * 4);
    float*          cbuf       = (float*)         alloc((size_t)BB * DEC_D * 4);
    unsigned short* h_bf       = (unsigned short*)alloc((size_t)BB * DEC_D * 2);
    unsigned short* hnew_bf    = (unsigned short*)alloc((size_t)BB * DEC_D * 2);
    (void)ws_size; (void)in_sizes; (void)n_in; (void)out_size;

    // ---- setup ----
    sort_kernel<<<1, 256, 0, stream>>>(cap_lens, enc_caps, sort_ind, len_sorted, cap_sorted);
    misc_out<<<(BB * TT + 2 * BB + 255) / 256, 256, 0, stream>>>(cap_sorted, len_sorted, sort_ind, out);

    auto conv = [&](const float* s, unsigned short* d, long long n_src, long long n_total) {
        f2bf_kernel<<<(unsigned)((n_total + 255) / 256), 256, 0, stream>>>(s, d, n_src, n_total);
    };
    conv(W_enc_att, Wenc_bf, (long long)ATT_D * OBJ_D, (long long)ATT_D * OBJ_D);
    conv(W_dec_att, Wdec_bf, (long long)ATT_D * DEC_D, (long long)ATT_D * DEC_D);
    conv(W_fbeta,   Wfb_bf,  (long long)OBJ_D * DEC_D, (long long)OBJ_D * DEC_D);
    conv(W_fc,      Wfc_bf,  (long long)VOCAB * DEC_D, (long long)VOCABP * DEC_D); // zero pad rows
    build_wcat<<<(unsigned)((4096LL * 4096 + 255) / 256), 256, 0, stream>>>(W_ih, W_hh, Wcat_bf);
    add_bias2<<<(4096 + 255) / 256, 256, 0, stream>>>(b_ih, b_hh, bcat, 4096);
    gather_obj<<<(unsigned)(((long long)BB * N_OBJ * OBJ_D + 255) / 256), 256, 0, stream>>>(objects, sort_ind, obj_bf);
    build_stepx<<<(unsigned)(((long long)BB * TT * EMB_D + 255) / 256), 256, 0, stream>>>(
        encoder_out, emb_table, sort_ind, cap_sorted, stepx_bf);
    init_hc<<<(BB * DEC_D + 255) / 256, 256, 0, stream>>>(hbuf, cbuf, h_bf);

    // att1 = objects_s @ W_enc_att^T + b : M=9216, N=1024, K=2048
    launch_gemm(obj_bf, OBJ_D, Wenc_bf, OBJ_D, b_enc_att, att1, ATT_D,
                BB * N_OBJ, ATT_D, ATT_D, OBJ_D, nullptr, 0, stream);

    // ---- sequential decode ----
    for (int t = 0; t < TT; ++t) {
        // att2 = h @ W_dec_att^T + b
        launch_gemm(h_bf, DEC_D, Wdec_bf, DEC_D, b_dec_att, att2, ATT_D,
                    BB, ATT_D, ATT_D, DEC_D, nullptr, 0, stream);
        // e[b,n]
        att_score<<<(BB * N_OBJ * 32 + 255) / 256, 256, 0, stream>>>(
            att1, att2, w_full_att, b_full_att, evec);
        // alpha + masked alphas
        softmax_alpha<<<1, 256, 0, stream>>>(evec, alpha, out + ALPHA_OFF, len_sorted, t);
        // fbeta gate (pre-sigmoid): h @ W_fbeta^T + b
        launch_gemm(h_bf, DEC_D, Wfb_bf, DEC_D, b_fbeta, fgate, OBJ_D,
                    BB, OBJ_D, OBJ_D, DEC_D, nullptr, 0, stream);
        // xcat = [x_t, sigmoid(gate)*awe, h]
        awe_gate_xcat<<<(BB * 4096 + 255) / 256, 256, 0, stream>>>(
            alpha, obj_bf, fgate, stepx_bf, h_bf, xh_bf, t);
        // gates = xcat @ [W_ih|W_hh]^T + (b_ih+b_hh) : M=256, N=4096, K=4096
        launch_gemm(xh_bf, 4096, Wcat_bf, 4096, bcat, gatesum, 4096,
                    BB, 4096, 4096, 4096, nullptr, 0, stream);
        // LSTM pointwise, masked state update
        lstm_pw<<<(BB * DEC_D + 255) / 256, 256, 0, stream>>>(
            gatesum, hbuf, cbuf, h_bf, hnew_bf, len_sorted, t);
        // preds = h_new @ W_fc^T + b_fc, masked, scattered into (B,T,VOCAB)
        launch_gemm(hnew_bf, DEC_D, Wfc_bf, DEC_D, b_fc,
                    out + PRED_OFF + (long long)t * VOCAB, (long long)TT * VOCAB,
                    BB, VOCAB, VOCABP, DEC_D, len_sorted, t, stream);
    }
}
